// NIFTY_47991964565962
// MI455X (gfx1250) — compile-verified, matrix operand order checked by reading, not run on hardware
//
#include <hip/hip_runtime.h>
#include <hip/hip_bf16.h>
#include <stdint.h>

#define N_NODES 100000
#define N_EDGES 3200000
#define F_IN    256
#define F_OUT   16

typedef __attribute__((ext_vector_type(16))) _Float16 v16h;
typedef __attribute__((ext_vector_type(8)))  float    v8f;

// ---------------------------------------------------------------------------
// K1: zero output, deg = 1.0 (self-loop contribution)
// ---------------------------------------------------------------------------
__global__ void gcn_init_kernel(float* __restrict__ out, float* __restrict__ deg) {
    long long t = (long long)blockIdx.x * blockDim.x + threadIdx.x;
    if (t < (long long)N_NODES * F_OUT) out[t] = 0.0f;
    if (t < N_NODES) deg[t] = 1.0f;
}

// ---------------------------------------------------------------------------
// K2: degree count over target nodes (col = edge_index[1])
// ---------------------------------------------------------------------------
__global__ void gcn_degree_kernel(const long long* __restrict__ edge_index,
                                  float* __restrict__ deg) {
    long long e = (long long)blockIdx.x * blockDim.x + threadIdx.x;
    if (e < N_EDGES) {
        int c = (int)edge_index[(long long)N_EDGES + e];
        atomicAdd(&deg[c], 1.0f);
    }
}

// ---------------------------------------------------------------------------
// K3: dinv = rsqrt(deg)   (deg >= 1 always, since self-loops are included)
// ---------------------------------------------------------------------------
__global__ void gcn_dinv_kernel(const float* __restrict__ deg,
                                float* __restrict__ dinv) {
    long long t = (long long)blockIdx.x * blockDim.x + threadIdx.x;
    if (t < N_NODES) dinv[t] = __frsqrt_rn(deg[t]);
}

// ---------------------------------------------------------------------------
// K4: xw = x @ W^T via v_wmma_f32_16x16x32_f16.
// One wave per 16-row tile of x; N = F_OUT = 16 is a single WMMA tile wide.
// K loop: 256 / 32 = 8 WMMA issues per tile, f32 accumulation.
//
// ISA 7.12.2 layouts (wave32):
//  A (16x32 f16): lane l -> row m = l&15; half = l>>4 selects K-halves.
//    vgpr i in 0..3 : K = k0 + half*8 + 2i,2i+1
//    vgpr i in 4..7 : K = k0 + 16 + half*8 + 2(i-4), +1
//  B (32x16 f16): same striping with column n = l&15; B[k][n] = W[n][k].
//  C/D (16x16 f32): lane l, vgpr j -> (m = (l>>4)*8 + j, n = l&15)
// ---------------------------------------------------------------------------
__global__ void __launch_bounds__(128)
gcn_wmma_gemm_kernel(const float* __restrict__ x,
                     const float* __restrict__ W,
                     float* __restrict__ xw) {
    const int wave = threadIdx.x >> 5;
    const int tile = blockIdx.x * 4 + wave;                 // 16-row tile index
    if (tile >= (N_NODES / 16)) return;                     // wave-uniform exit

    const int lane = threadIdx.x & 31;
    const int half = lane >> 4;                             // 0 or 1
    const int m    = lane & 15;                             // row-in-tile / col

    const int row = tile * 16 + m;                          // node row for A
    const float2* __restrict__ xrow = (const float2*)(x + (size_t)row * F_IN);
    const float2* __restrict__ wrow = (const float2*)(W + (size_t)m   * F_IN);

    v8f acc = {0.f, 0.f, 0.f, 0.f, 0.f, 0.f, 0.f, 0.f};

    #pragma unroll
    for (int kk = 0; kk < F_IN / 32; ++kk) {
        const int k0 = kk * 32;
        v16h a, b;
        #pragma unroll
        for (int i = 0; i < 8; ++i) {
            const int kb = k0 + ((i < 4) ? (half * 8 + 2 * i)
                                         : (16 + half * 8 + 2 * (i - 4)));
            const float2 va = xrow[kb >> 1];
            const float2 vb = wrow[kb >> 1];
            a[2 * i]     = (_Float16)va.x;
            a[2 * i + 1] = (_Float16)va.y;
            b[2 * i]     = (_Float16)vb.x;
            b[2 * i + 1] = (_Float16)vb.y;
        }
        acc = __builtin_amdgcn_wmma_f32_16x16x32_f16(
            /*neg_a=*/false, a, /*neg_b=*/false, b,
            /*c_mod=*/(short)0, acc, /*reuse_a=*/false, /*reuse_b=*/false);
    }

    // Write back D: lane holds column n=m, rows (half*8 + j), j = 0..7
    #pragma unroll
    for (int j = 0; j < 8; ++j) {
        const int out_row = tile * 16 + half * 8 + j;
        xw[(size_t)out_row * F_OUT + m] = acc[j];
    }
}

// ---------------------------------------------------------------------------
// K5: edge scatter: out[col] += xw[row] * dinv[row]*dinv[col]
// One thread per (edge, feature). xw/out are L2-resident (6.4 MB each).
// ---------------------------------------------------------------------------
__global__ void gcn_scatter_kernel(const long long* __restrict__ edge_index,
                                   const float* __restrict__ xw,
                                   const float* __restrict__ dinv,
                                   float* __restrict__ out) {
    long long t = (long long)blockIdx.x * blockDim.x + threadIdx.x;
    long long e = t >> 4;
    int f = (int)(t & 15);
    if (e >= N_EDGES) return;
    int r = (int)edge_index[e];
    int c = (int)edge_index[(long long)N_EDGES + e];
    float norm = dinv[r] * dinv[c];
    atomicAdd(&out[(size_t)c * F_OUT + f], xw[(size_t)r * F_OUT + f] * norm);
}

// ---------------------------------------------------------------------------
// K6: self-loop term + bias: out[n] += xw[n] * dinv[n]^2 + b
// ---------------------------------------------------------------------------
__global__ void gcn_finalize_kernel(const float* __restrict__ xw,
                                    const float* __restrict__ dinv,
                                    const float* __restrict__ bias,
                                    float* __restrict__ out) {
    long long t = (long long)blockIdx.x * blockDim.x + threadIdx.x;
    if (t >= (long long)N_NODES * F_OUT) return;
    int n = (int)(t >> 4);
    int f = (int)(t & 15);
    float di = dinv[n];
    out[t] += xw[t] * di * di + bias[f];
}

// ---------------------------------------------------------------------------
extern "C" void kernel_launch(void* const* d_in, const int* in_sizes, int n_in,
                              void* d_out, int out_size, void* d_ws, size_t ws_size,
                              hipStream_t stream) {
    const float*      x  = (const float*)d_in[0];
    const long long*  ei = (const long long*)d_in[1];   // int64 [2, E]
    const float*      W  = (const float*)d_in[2];       // [F_OUT, F_IN]
    const float*      b  = (const float*)d_in[3];       // [F_OUT]
    float*            out = (float*)d_out;              // [N, F_OUT]

    // Workspace layout: deg | dinv | xw
    float* deg  = (float*)d_ws;
    float* dinv = deg + N_NODES;
    float* xw   = dinv + N_NODES;                       // N * F_OUT floats

    const int B = 256;

    // K1: init out + deg
    {
        long long n = (long long)N_NODES * F_OUT;
        gcn_init_kernel<<<(unsigned)((n + B - 1) / B), B, 0, stream>>>(out, deg);
    }
    // K2: degree
    gcn_degree_kernel<<<(N_EDGES + B - 1) / B, B, 0, stream>>>(ei, deg);
    // K3: rsqrt
    gcn_dinv_kernel<<<(N_NODES + B - 1) / B, B, 0, stream>>>(deg, dinv);
    // K4: WMMA GEMM, 4 waves (tiles) per 128-thread block
    {
        int tiles = N_NODES / 16;                       // 6250
        int blocks = (tiles + 3) / 4;
        gcn_wmma_gemm_kernel<<<blocks, 128, 0, stream>>>(x, W, xw);
    }
    // K5: edge scatter-add
    {
        long long n = (long long)N_EDGES * F_OUT;
        gcn_scatter_kernel<<<(unsigned)((n + B - 1) / B), B, 0, stream>>>(ei, xw, dinv, out);
    }
    // K6: self loops + bias
    {
        long long n = (long long)N_NODES * F_OUT;
        gcn_finalize_kernel<<<(unsigned)((n + B - 1) / B), B, 0, stream>>>(xw, dinv, b, out);
    }
}
